// PolicyNetwork_63780264345838
// MI455X (gfx1250) — matrix operand order repaired and checked by reading, not compiled, not run on hardware
//
#include <hip/hip_runtime.h>
#include <hip/hip_bf16.h>

// ---------------------------------------------------------------------------
// PolicyNetwork fused kernel for gfx1250 (MI455X)
//   B=16384, OBS=128, H=256, DEPTH=128, A=32
// ---------------------------------------------------------------------------

typedef __attribute__((ext_vector_type(16))) _Float16 v16h;
typedef __attribute__((ext_vector_type(8)))  _Float16 v8h;
typedef __attribute__((ext_vector_type(8)))  float    v8f;
typedef __attribute__((ext_vector_type(4)))  float    f4;

constexpr int BTOT = 16384;
constexpr int OBSN = 128;
constexpr int HD   = 256;
constexpr int DEP  = 128;
constexpr int ACT  = 32;
constexpr int K1   = OBSN + HD;   // 384
constexpr int KT1  = K1 / 32;     // 12 k-tiles for GEMM1
constexpr int KT2  = HD / 32;     // 8 k-tiles for GEMM2
constexpr int AS   = 392;         // Asub row stride in f16 (mult of 8 -> 16B rows)
constexpr int HS   = 264;         // Hbuf row stride in f16
constexpr int PS   = 260;         // P32 row stride in f32

// ---------------------------------------------------------------------------
// Prep: swizzle f32 row-major weight W[k][n] (n-cols = 256) into f16 WMMA
// B-operand tiles. Tile (kt,nt) is 512 f16 = [lane 0..31][j 0..15], where
// lane holds column n = nt*16 + lane%16 and slot j holds k = kt*32 + 8*(lane/16) + (j%8) + 16*(j/8).
// ---------------------------------------------------------------------------
__global__ __launch_bounds__(256)
void swizzle_w(const float* __restrict__ W, _Float16* __restrict__ out, int KT) {
    int e = blockIdx.x * 256 + threadIdx.x;
    int total = KT * 16 * 512;
    if (e >= total) return;
    int tile = e >> 9;
    int r    = e & 511;
    int lane = r >> 4;
    int j    = r & 15;
    int kt   = tile >> 4;
    int nt   = tile & 15;
    int n    = (nt << 4) + (lane & 15);
    int k    = kt * 32 + ((lane >> 4) << 3) + (j & 7) + ((j >> 3) << 4);
    out[e] = (_Float16)W[(size_t)k * 256 + n];
}

// ---------------------------------------------------------------------------
// 2 GiB streaming copy: new_buf = stack_buf (non-temporal, float4)
// ---------------------------------------------------------------------------
__global__ __launch_bounds__(256)
void copy_buf(const f4* __restrict__ src, f4* __restrict__ dst, long long n4) {
    long long i      = (long long)blockIdx.x * 256 + threadIdx.x;
    long long stride = (long long)gridDim.x * 256;
    for (; i < n4; i += stride) {
        f4 v = __builtin_nontemporal_load(src + i);
        __builtin_nontemporal_store(v, dst + i);
    }
}

// ---------------------------------------------------------------------------
// Fused forward: gather top, GEMM1+tanh, GEMM2+tanh (WMMA f16->f32), heads,
// argmax/push/pop, logits, value, new_ptr, and overwrite pushed rows in out_buf.
// One block = 16 batch rows. 128 threads = 4 waves; wave w owns N-tiles 4w..4w+3.
// ---------------------------------------------------------------------------
__global__ __launch_bounds__(128)
void policy_fused(const float* __restrict__ x,
                  const float* __restrict__ stack_buf,
                  const int*   __restrict__ stack_ptr,
                  const _Float16* __restrict__ wsW1,
                  const _Float16* __restrict__ wsW2,
                  const float* __restrict__ b1, const float* __restrict__ b2,
                  const float* __restrict__ Ws, const float* __restrict__ bs,
                  const float* __restrict__ Wp, const float* __restrict__ bp,
                  const float* __restrict__ Wv, const float* __restrict__ bv,
                  float* __restrict__ out_logits, float* __restrict__ out_value,
                  float* __restrict__ out_buf,    float* __restrict__ out_ptr) {
    __shared__ __align__(16) _Float16 Asub[16 * AS];   // [x | top] as f16
    __shared__ __align__(16) _Float16 Hbuf[16 * HS];   // h as f16 (A of GEMM2)
    __shared__ __align__(16) float    P32[16 * PS];    // p as f32 (heads + push rows)
    __shared__ float Sbuf[16 * 3];
    __shared__ int   sptr[16];
    __shared__ int   pushIdx[16];

    const int tid  = threadIdx.x;
    const int lane = tid & 31;
    const int wave = tid >> 5;
    const int half = lane >> 4;
    const int lmod = lane & 15;
    const size_t m0 = (size_t)blockIdx.x * 16;

    if (tid < 16) sptr[tid] = stack_ptr[m0 + tid];
    __syncthreads();

    // ---- stage A = [x | top-of-stack] in f16 -------------------------------
    for (int i = tid; i < 16 * OBSN; i += 128) {
        int m = i >> 7, c = i & (OBSN - 1);
        Asub[m * AS + c] = (_Float16)x[(m0 + m) * OBSN + c];
    }
    for (int i = tid; i < 16 * HD; i += 128) {
        int m = i >> 8, c = i & (HD - 1);
        const float* row = stack_buf + ((m0 + m) * (size_t)DEP + (size_t)(sptr[m] - 1)) * HD;
        Asub[m * AS + OBSN + c] = (_Float16)row[c];
    }
    __syncthreads();

    // ---- GEMM1: h = tanh([x|top] @ W1 + b1) --------------------------------
    for (int nt = 0; nt < 4; ++nt) {
        const int ntile = wave * 4 + nt;
        v8f acc = {};
        for (int kt = 0; kt < KT1; ++kt) {
            const _Float16* ap = &Asub[lmod * AS + kt * 32 + half * 8];
            v8h alo = *(const v8h*)ap;
            v8h ahi = *(const v8h*)(ap + 16);
            v16h a  = __builtin_shufflevector(alo, ahi, 0,1,2,3,4,5,6,7,8,9,10,11,12,13,14,15);
            v16h b  = *(const v16h*)(wsW1 + (((size_t)(kt * 16 + ntile)) << 9) + (lane << 4));
            acc = __builtin_amdgcn_wmma_f32_16x16x32_f16(false, a, false, b,
                                                         (short)0, acc, false, false);
        }
        const int   n    = ntile * 16 + lmod;
        const float bias = b1[n];
#pragma unroll
        for (int r = 0; r < 8; ++r) {
            float hv = tanhf(acc[r] + bias);
            Hbuf[(r + half * 8) * HS + n] = (_Float16)hv;
        }
    }
    __syncthreads();

    // ---- GEMM2: p = tanh(h @ W2 + b2) --------------------------------------
    for (int nt = 0; nt < 4; ++nt) {
        const int ntile = wave * 4 + nt;
        v8f acc = {};
        for (int kt = 0; kt < KT2; ++kt) {
            const _Float16* ap = &Hbuf[lmod * HS + kt * 32 + half * 8];
            v8h alo = *(const v8h*)ap;
            v8h ahi = *(const v8h*)(ap + 16);
            v16h a  = __builtin_shufflevector(alo, ahi, 0,1,2,3,4,5,6,7,8,9,10,11,12,13,14,15);
            v16h b  = *(const v16h*)(wsW2 + (((size_t)(kt * 16 + ntile)) << 9) + (lane << 4));
            acc = __builtin_amdgcn_wmma_f32_16x16x32_f16(false, a, false, b,
                                                         (short)0, acc, false, false);
        }
        const int   n    = ntile * 16 + lmod;
        const float bias = b2[n];
#pragma unroll
        for (int r = 0; r < 8; ++r)
            P32[(r + half * 8) * PS + n] = tanhf(acc[r] + bias);
    }
    __syncthreads();

    // ---- heads: 8 threads per row; thread j does logits {j,j+8,j+16,j+24},
    //      j<3 also the stack-op logit j, j==3 also the value head. ----------
    {
        const int hm = tid >> 3;
        const int hj = tid & 7;
        float accL0 = 0.f, accL1 = 0.f, accL2 = 0.f, accL3 = 0.f;
        float accS = 0.f, accV = 0.f;
        const float* prow = &P32[hm * PS];
        for (int k = 0; k < HD; ++k) {
            float pv = prow[k];
            const float* wk = &Wp[k * ACT + hj];
            accL0 += pv * wk[0];
            accL1 += pv * wk[8];
            accL2 += pv * wk[16];
            accL3 += pv * wk[24];
            if (hj < 3)        accS += pv * Ws[k * 3 + hj];
            else if (hj == 3)  accV += pv * Wv[k];
        }
        float* lo = &out_logits[(m0 + hm) * ACT + hj];
        lo[0]  = accL0 + bp[hj];
        lo[8]  = accL1 + bp[hj + 8];
        lo[16] = accL2 + bp[hj + 16];
        lo[24] = accL3 + bp[hj + 24];
        if (hj < 3)       Sbuf[hm * 3 + hj] = accS + bs[hj];
        else if (hj == 3) out_value[m0 + hm] = accV + bv[0];
    }
    __syncthreads();

    // ---- stack-op decision (softmax argmax == logit argmax, first-max) -----
    if (tid < 16) {
        float s0 = Sbuf[tid * 3 + 0], s1 = Sbuf[tid * 3 + 1], s2 = Sbuf[tid * 3 + 2];
        int op = 0; float best = s0;
        if (s1 > best) { best = s1; op = 1; }
        if (s2 > best) { op = 2; }
        int ptr  = sptr[tid];
        int push = (op == 2 && ptr < DEP) ? 1 : 0;
        int pop  = (op == 1 && ptr > 1)   ? 1 : 0;
        pushIdx[tid] = push ? ptr : -1;   // safe_idx == ptr when pushing
        out_ptr[m0 + tid] = (float)(ptr + push - pop);
    }
    __syncthreads();

    // ---- overwrite pushed rows in new_buf (copy kernel wrote the rest) -----
    for (int m = 0; m < 16; ++m) {
        int si = pushIdx[m];
        if (si >= 0) {
            for (int c = tid; c < HD; c += 128)
                out_buf[((m0 + m) * (size_t)DEP + si) * HD + c] = P32[m * PS + c];
        }
    }
}

// ---------------------------------------------------------------------------
extern "C" void kernel_launch(void* const* d_in, const int* in_sizes, int n_in,
                              void* d_out, int out_size, void* d_ws, size_t ws_size,
                              hipStream_t stream) {
    const float* x         = (const float*)d_in[0];
    const float* stack_buf = (const float*)d_in[1];
    const int*   stack_ptr = (const int*)  d_in[2];
    const float* W1        = (const float*)d_in[3];
    const float* b1        = (const float*)d_in[4];
    const float* W2        = (const float*)d_in[5];
    const float* b2        = (const float*)d_in[6];
    const float* Ws        = (const float*)d_in[7];
    const float* bs        = (const float*)d_in[8];
    const float* Wp        = (const float*)d_in[9];
    const float* bp        = (const float*)d_in[10];
    const float* Wv        = (const float*)d_in[11];
    const float* bv        = (const float*)d_in[12];

    float* out        = (float*)d_out;
    float* out_logits = out;                                        // [B, 32]
    float* out_value  = out_logits + (size_t)BTOT * ACT;            // [B]
    float* out_buf    = out_value + BTOT;                           // [B, DEPTH, H]
    float* out_ptr    = out_buf + (size_t)BTOT * DEP * HD;          // [B] (as f32)

    _Float16* wsW1 = (_Float16*)d_ws;                 // 12*16*512 = 98304 f16
    _Float16* wsW2 = wsW1 + (size_t)KT1 * 16 * 512;   //  8*16*512 = 65536 f16

    // 1) weight swizzle (tiny)
    swizzle_w<<<(KT1 * 16 * 512 + 255) / 256, 256, 0, stream>>>(W1, wsW1, KT1);
    swizzle_w<<<(KT2 * 16 * 512 + 255) / 256, 256, 0, stream>>>(W2, wsW2, KT2);

    // 2) bulk new_buf = stack_buf (2 GiB, bandwidth-bound => NT float4 stream)
    long long n4 = (long long)BTOT * DEP * HD / 4;
    copy_buf<<<16384, 256, 0, stream>>>((const f4*)stack_buf, (f4*)out_buf, n4);

    // 3) fused forward; overwrites only pushed rows (stream-ordered after copy)
    policy_fused<<<BTOT / 16, 128, 0, stream>>>(x, stack_buf, stack_ptr, wsW1, wsW2,
                                                b1, b2, Ws, bs, Wp, bp, Wv, bv,
                                                out_logits, out_value, out_buf, out_ptr);
}
